// PaiConv_6597069766757
// MI455X (gfx1250) — matrix operand order repaired and checked by reading, not compiled, not run on hardware
//
#include <hip/hip_runtime.h>

typedef __attribute__((ext_vector_type(16))) __bf16 v16bf;
typedef __attribute__((ext_vector_type(8)))  float  v8f;
typedef __attribute__((ext_vector_type(4)))  unsigned int v4u;

#define BSZ   4
#define NPTS  8192
#define KNB   32
#define CIN   64
#define COUT  64
#define TOTPTS (BSZ*NPTS)      // 32768
#define FLATK  3072            // 96*32

// workspace byte offsets
#define WS_WC    0u            // W_conv packed: 4*96*32*16 ushort = 393216 B
#define WS_WM    393216u       // W_mlp packed: 2*2*32*16 ushort = 4096 B
#define WS_STATS 397312u       // 128 floats
#define WS_FEATT 401408u       // featT bf16: 32768*64 ushort = 4 MiB

// dynamic LDS byte offsets (per block)
#define L_BMAT  0       // 7*32 f32 = 896
#define L_KERN  896     // 3*32 f32 = 384
#define L_PAD   1280    // 32*32 f32 = 4096
#define L_PERM  5632    // per-wave 32*32 bf16 = 2048  (x4 waves)
#define L_PE    13824   // per-wave 32*64 bf16 = 4096  (x4 waves)
#define L_FEATS 30208   // per-wave 96*32 bf16 = 6144  (x4 waves)
#define L_FLATG 54784   // 16 pts * 3072 bf16 = 98304
#define L_TOTAL 153088

union Frag { v16bf bf; v4u u[2]; };

static __device__ __forceinline__ unsigned short f2bf(float f) {
    union { __bf16 h; unsigned short s; } v;
    v.h = (__bf16)f;               // hardware RNE f32->bf16 conversion
    return v.s;
}
static __device__ __forceinline__ unsigned int pk2bf(float lo, float hi) {
    union { __bf16 h[2]; unsigned int u; } v;
    v.h[0] = (__bf16)lo; v.h[1] = (__bf16)hi;
    return v.u;
}

// ---------- kernel 0a: feature (B,64,N) f32 -> featT (B*N,64) bf16 ----------
__global__ void featT_kernel(const float* __restrict__ feature,
                             unsigned short* __restrict__ featT) {
    int gp = blockIdx.x * blockDim.x + threadIdx.x;       // 0..32767
    int b = gp >> 13, n = gp & (NPTS - 1);
    const float* src = feature + (size_t)b * CIN * NPTS + n;
    unsigned int* d32 = (unsigned int*)(featT + (size_t)gp * CIN);
#pragma unroll
    for (int j = 0; j < 32; ++j)
        d32[j] = pk2bf(src[(size_t)(2 * j) * NPTS], src[(size_t)(2 * j + 1) * NPTS]);
}

// ---------- kernel 0b: pack W_conv / W_mlp into WMMA A-operand layout ----------
// A fragment order per lane (16 elems): lane<16: K = {e (e<8), e+8 (e>=8)}; lane>=16: +8
__global__ void pack_kernel(const float* __restrict__ Wc, const float* __restrict__ Wm,
                            unsigned short* __restrict__ wcp, unsigned short* __restrict__ wmp) {
    int idx = blockIdx.x * blockDim.x + threadIdx.x;
    const int NWC = 4 * 96 * 32 * 16;
    if (idx < NWC) {
        int e = idx & 15, lane = (idx >> 4) & 31, t = idx >> 9;
        int ks = t % 96, mt = t / 96;
        int half = lane >> 4, m = lane & 15;
        int kl = (e < 8 ? e : e + 8) + half * 8;
        wcp[idx] = f2bf(Wc[(size_t)(mt * 16 + m) * FLATK + ks * 32 + kl]);
    } else {
        int j = idx - NWC;
        if (j < 2 * 2 * 32 * 16) {
            int e = j & 15, lane = (j >> 4) & 31, t = j >> 9;
            int ks = t & 1, mt = t >> 1;
            int half = lane >> 4, m = lane & 15;
            int kl = (e < 8 ? e : e + 8) + half * 8;
            wmp[j] = f2bf(Wm[(mt * 16 + m) * 64 + ks * 32 + kl]);
        }
    }
}

// ---------- kernel 1: main PaiConv compute (pre-BN), WMMA bf16 ----------
__global__ void __launch_bounds__(128) paiconv_main(
    const float* __restrict__ x, const int* __restrict__ neigh,
    const float* __restrict__ Bmat, const float* __restrict__ kern,
    const float* __restrict__ onepad, const float* __restrict__ b_mlp,
    const float* __restrict__ b_conv,
    const unsigned short* __restrict__ featT,
    const unsigned short* __restrict__ wcp, const unsigned short* __restrict__ wmp,
    float* __restrict__ out) {
    extern __shared__ char smem[];
    float* sB = (float*)(smem + L_BMAT);
    float* sK = (float*)(smem + L_KERN);
    float* sP = (float*)(smem + L_PAD);

    const int tid = threadIdx.x, wave = tid >> 5, lane = tid & 31;
    const int lhalf = lane >> 4, lmod = lane & 15;

    for (int i = tid; i < 7 * 32; i += 128) sB[i] = Bmat[i];
    for (int i = tid; i < 3 * 32; i += 128) sK[i] = kern[i];
    for (int i = tid; i < 32 * 32; i += 128) sP[i] = onepad[i];
    __syncthreads();

    char* permT  = smem + L_PERM  + wave * 2048;   // [j][k] bf16 (relu'd, unnormalized)
    char* peL    = smem + L_PE    + wave * 4096;   // [k][f] bf16
    char* featsL = smem + L_FEATS + wave * 6144;   // [ch][k] bf16
    char* flatG  = smem + L_FLATG;                 // [pt][ch*32+j] bf16

    // loop-invariant preloads: W_mlp A-fragments + b_mlp biases
    Frag wmA[4];
#pragma unroll
    for (int q = 0; q < 4; ++q) {
        const v4u* ap = (const v4u*)(wmp + (size_t)(q * 32 + lane) * 16);
        wmA[q].u[0] = ap[0]; wmA[q].u[1] = ap[1];
    }
    float bm[16];
#pragma unroll
    for (int mt = 0; mt < 2; ++mt)
#pragma unroll
        for (int rI = 0; rI < 8; ++rI)
            bm[mt * 8 + rI] = b_mlp[mt * 16 + rI + lhalf * 8];

    // all-ones A fragment for column-sum WMMA
    Frag ones;
    {
        v4u o = { 0x3F803F80u, 0x3F803F80u, 0x3F803F80u, 0x3F803F80u };
        ones.u[0] = o; ones.u[1] = o;
    }

    for (int pt = 0; pt < 4; ++pt) {
        const int p  = wave * 4 + pt;              // point in tile
        const int gp = blockIdx.x * 16 + p;        // global point
        const int b = gp >> 13;
        const int idx = neigh[(size_t)gp * KNB + lane];   // lane = neighbor k
        const float* xb = x + (size_t)b * 3 * NPTS;
        float px = xb[idx], py = xb[NPTS + idx], pz = xb[2 * NPTS + idx];
        float rx = __shfl(px, 0, 32), ry = __shfl(py, 0, 32), rz = __shfl(pz, 0, 32);
        float dx = px - rx, dy = py - ry, dz = pz - rz;
        float dis = sqrtf(fmaxf(dx * dx + dy * dy + dz * dz, 1e-12f));
        const float TP = 6.283185307179586f;
        float rff[7] = { TP * rx, TP * ry, TP * rz, TP * dx, TP * dy, TP * dz, TP * dis };

        // feature gather: pipeline all 8 loads, then scatter to featsL rows 0..63
        {
            const v4u* fsrc = (const v4u*)(featT + (size_t)(b * NPTS + idx) * CIN);
            v4u fv[8];
#pragma unroll
            for (int q = 0; q < 8; ++q) fv[q] = fsrc[q];
#pragma unroll
            for (int q = 0; q < 8; ++q)
#pragma unroll
                for (int w2 = 0; w2 < 4; ++w2) {
                    unsigned int u = fv[q][w2];
                    int ch = q * 8 + w2 * 2;
                    *(unsigned short*)(featsL + ch * 64 + lane * 2)       = (unsigned short)(u & 0xffff);
                    *(unsigned short*)(featsL + (ch + 1) * 64 + lane * 2) = (unsigned short)(u >> 16);
                }
        }

        // pe row for this k: sin(proj_j) at f=j, cos at f=32+j, packed bf16
        unsigned int* peRow = (unsigned int*)(peL + lane * 128);
#pragma unroll
        for (int jj = 0; jj < 16; ++jj) {
            float p0 = 0.f, p1 = 0.f;
#pragma unroll
            for (int f = 0; f < 7; ++f) {
                p0 += rff[f] * sB[f * 32 + 2 * jj];
                p1 += rff[f] * sB[f * 32 + 2 * jj + 1];
            }
            peRow[jj]      = pk2bf(__sinf(p0), __sinf(p1));
            peRow[16 + jj] = pk2bf(__cosf(p0), __cosf(p1));
        }

        // perm row (lane = k): relu(x_rel·kernels + one_padding); normalization
        // deferred to stage 2 via ones-WMMA column sums
#pragma unroll
        for (int j = 0; j < 32; ++j) {
            float v = dx * sK[j] + dy * sK[32 + j] + dz * sK[64 + j] + sP[lane * 32 + j];
            *(unsigned short*)(permT + j * 64 + lane * 2) = f2bf(fmaxf(v, 0.f));
        }

        // stage 1b: x_feats(32x32) = W_mlp(32x64) @ peT -> feats rows 64..95
#pragma unroll
        for (int mt = 0; mt < 2; ++mt)
#pragma unroll
            for (int kt = 0; kt < 2; ++kt) {
                v8f acc = {};
#pragma unroll
                for (int ks = 0; ks < 2; ++ks) {
                    Frag Bv;
                    const char* bp = peL + (kt * 16 + lmod) * 128 + ks * 64 + lhalf * 16;
                    Bv.u[0] = *(const v4u*)bp; Bv.u[1] = *(const v4u*)(bp + 32);
                    acc = __builtin_amdgcn_wmma_f32_16x16x32_bf16(false, wmA[mt * 2 + ks].bf,
                                                                  false, Bv.bf,
                                                                  (short)0, acc, false, false);
                }
#pragma unroll
                for (int rI = 0; rI < 8; ++rI) {
                    int m = rI + lhalf * 8;
                    float v = acc[rI] + bm[mt * 8 + rI];
                    *(unsigned short*)(featsL + (64 + mt * 16 + m) * 64 + (kt * 16 + lmod) * 2) = f2bf(v);
                }
            }

        // stage 2: G(96x32) = feats(96x32) @ relu_perm(32x32), column-scaled by
        // 1/(colsum+1e-6) computed with a ones-row WMMA -> flatG[p]
        char* fg = flatG + p * 6144;
#pragma unroll
        for (int jt = 0; jt < 2; ++jt) {
            Frag Bv;
            const char* bp = permT + (jt * 16 + lmod) * 64 + lhalf * 32;
            Bv.u[0] = *(const v4u*)bp; Bv.u[1] = *(const v4u*)(bp + 16);
            v8f cacc = {};
            cacc = __builtin_amdgcn_wmma_f32_16x16x32_bf16(false, ones.bf, false, Bv.bf,
                                                           (short)0, cacc, false, false);
            float rcp = __builtin_amdgcn_rcpf(cacc[0] + 1e-6f);  // colsum for this lane's j
#pragma unroll
            for (int mt = 0; mt < 6; ++mt) {
                Frag A;
                const char* ap = featsL + (mt * 16 + lmod) * 64 + lhalf * 32;
                A.u[0] = *(const v4u*)ap;
                A.u[1] = *(const v4u*)(ap + 16);
                v8f acc = {};
                acc = __builtin_amdgcn_wmma_f32_16x16x32_bf16(false, A.bf, false, Bv.bf,
                                                              (short)0, acc, false, false);
#pragma unroll
                for (int rI = 0; rI < 8; ++rI) {
                    int ch = mt * 16 + rI + lhalf * 8;
                    int j = jt * 16 + lmod;
                    *(unsigned short*)(fg + (ch * 32 + j) * 2) = f2bf(acc[rI] * rcp);
                }
            }
        }
    }
    __syncthreads();

    // stage 3: Out(64 x 16pts) = W_conv(64x3072) @ Flat(3072x16); wave = M tile
    {
        v8f acc = {};
        const unsigned short* wbase = wcp + (size_t)wave * 96 * 32 * 16;
#pragma unroll 4
        for (int ks = 0; ks < 96; ++ks) {
            Frag A, Bv;
            const v4u* ap = (const v4u*)(wbase + ((size_t)ks * 32 + lane) * 16);
            A.u[0] = ap[0]; A.u[1] = ap[1];
            const char* bp = flatG + lmod * 6144 + ks * 64 + lhalf * 16;
            Bv.u[0] = *(const v4u*)bp; Bv.u[1] = *(const v4u*)(bp + 32);
            acc = __builtin_amdgcn_wmma_f32_16x16x32_bf16(false, A.bf, false, Bv.bf,
                                                          (short)0, acc, false, false);
        }
        int gp = blockIdx.x * 16 + lmod;
        int b = gp >> 13, n = gp & (NPTS - 1);
#pragma unroll
        for (int rI = 0; rI < 8; ++rI) {
            int c = wave * 16 + rI + lhalf * 8;
            out[(size_t)(b * COUT + c) * NPTS + n] = acc[rI] + b_conv[c];
        }
    }
}

// ---------- kernel 2: per-channel mean/var -> scale/shift (deterministic tree) ----------
__global__ void stats_kernel(const float* __restrict__ out, const float* __restrict__ gamma,
                             const float* __restrict__ beta, float* __restrict__ stats) {
    __shared__ float ss[256], sq[256];
    int c = blockIdx.x;
    float s = 0.f, q = 0.f;
    for (int i = threadIdx.x; i < TOTPTS; i += 256) {
        int b = i >> 13, n = i & (NPTS - 1);
        float v = out[(size_t)(b * COUT + c) * NPTS + n];
        s += v; q += v * v;
    }
    ss[threadIdx.x] = s; sq[threadIdx.x] = q;
    __syncthreads();
    for (int st = 128; st > 0; st >>= 1) {
        if (threadIdx.x < st) {
            ss[threadIdx.x] += ss[threadIdx.x + st];
            sq[threadIdx.x] += sq[threadIdx.x + st];
        }
        __syncthreads();
    }
    if (threadIdx.x == 0) {
        float mean = ss[0] / (float)TOTPTS;
        float var  = sq[0] / (float)TOTPTS - mean * mean;
        float scale = gamma[c] * rsqrtf(var + 1e-5f);
        stats[c]      = scale;
        stats[64 + c] = beta[c] - mean * scale;
    }
}

// ---------- kernel 3: apply BN in place ----------
__global__ void bn_apply(float* __restrict__ out, const float* __restrict__ stats) {
    int i = blockIdx.x * 256 + threadIdx.x;     // layout (B, C, N)
    int c = (i >> 13) & 63;
    out[i] = out[i] * stats[c] + stats[64 + c];
}

extern "C" void kernel_launch(void* const* d_in, const int* in_sizes, int n_in,
                              void* d_out, int out_size, void* d_ws, size_t ws_size,
                              hipStream_t stream) {
    const float* x        = (const float*)d_in[0];
    const float* feature  = (const float*)d_in[1];
    const int*   neigh    = (const int*)  d_in[2];
    const float* Bmat     = (const float*)d_in[3];
    const float* kernels  = (const float*)d_in[4];
    const float* onepad   = (const float*)d_in[5];
    const float* W_mlp    = (const float*)d_in[6];
    const float* b_mlp    = (const float*)d_in[7];
    const float* W_conv   = (const float*)d_in[8];
    const float* b_conv   = (const float*)d_in[9];
    const float* gamma    = (const float*)d_in[10];
    const float* beta     = (const float*)d_in[11];
    float* out = (float*)d_out;
    char*  ws  = (char*)d_ws;

    unsigned short* wcp   = (unsigned short*)(ws + WS_WC);
    unsigned short* wmp   = (unsigned short*)(ws + WS_WM);
    float*          stats = (float*)         (ws + WS_STATS);
    unsigned short* featT = (unsigned short*)(ws + WS_FEATT);

    featT_kernel<<<TOTPTS / 256, 256, 0, stream>>>(feature, featT);
    pack_kernel<<<(4 * 96 * 32 * 16 + 2 * 2 * 32 * 16 + 255) / 256, 256, 0, stream>>>(
        W_conv, W_mlp, wcp, wmp);
    paiconv_main<<<TOTPTS / 16, 128, L_TOTAL, stream>>>(
        x, neigh, Bmat, kernels, onepad, b_mlp, b_conv, featT, wcp, wmp, out);
    stats_kernel<<<64, 256, 0, stream>>>(out, gamma, beta, stats);
    bn_apply<<<(TOTPTS * COUT) / 256, 256, 0, stream>>>(out, stats);
}
